// RecurrentAgent_2817498546571
// MI455X (gfx1250) — matrix operand order repaired and checked by reading, not compiled, not run on hardware
//
#include <hip/hip_runtime.h>
#include <hip/hip_bf16.h>

typedef __bf16 bf16;
typedef __attribute__((ext_vector_type(16))) __bf16 v16bf;
typedef __attribute__((ext_vector_type(8)))  __bf16 v8bf;
typedef __attribute__((ext_vector_type(8)))  float  v8f;

// ---------------- WMMA fragment loaders (CDNA5 wave32 layouts) ----------------
// A 16x32 bf16 from row-major LDS tile: lane r<16 holds row M=r;
// v0..3 -> K(half*8+0..7), v4..7 -> K(16+half*8+0..7): two contiguous 16B runs.
__device__ __forceinline__ v16bf frag_a(const bf16* Arow, int lda, int lane) {
  int r = lane & 15, half = lane >> 4;
  const bf16* p = Arow + r * lda + half * 8;
  v8bf lo = *(const v8bf*)(p);
  v8bf hi = *(const v8bf*)(p + 16);
  return __builtin_shufflevector(lo, hi, 0,1,2,3,4,5,6,7,8,9,10,11,12,13,14,15);
}
// B 32x16 bf16 from LDS stored TRANSPOSED [N][K]: lane n holds col N=n,
// v0..3 -> K(kh*8+0..7), v4..7 -> K(16+kh*8+0..7): two contiguous 16B runs.
__device__ __forceinline__ v16bf frag_bt(const bf16* Bcol, int ldk, int lane) {
  int n = lane & 15, kh = lane >> 4;
  const bf16* p = Bcol + n * ldk + kh * 8;
  v8bf lo = *(const v8bf*)(p);
  v8bf hi = *(const v8bf*)(p + 16);
  return __builtin_shufflevector(lo, hi, 0,1,2,3,4,5,6,7,8,9,10,11,12,13,14,15);
}

// ---------------- CDNA5 async global->LDS copy (ASYNCcnt path) ----------------
__device__ __forceinline__ void async_copy_b64(void* lds_ptr, const void* gaddr) {
  unsigned lds_off = (unsigned)(size_t)lds_ptr;   // LDS aperture: low 32 bits = LDS addr
  asm volatile("global_load_async_to_lds_b64 %0, %1, off"
               :: "v"(lds_off), "v"((unsigned long long)(size_t)gaddr) : "memory");
}
__device__ __forceinline__ void async_wait0() {
  asm volatile("s_wait_asynccnt 0x0" ::: "memory");
}

// ---------------- weight prep ----------------
// generic: f32 [N][K] -> bf16 [K][N] (c-major k, used for conv1 / w_ih / w_hh)
__global__ void prep_wt(const float* __restrict__ w, bf16* __restrict__ wt, int N, int K) {
  int i = blockIdx.x * 256 + threadIdx.x;
  if (i >= N * K) return;
  int n = i / K, k = i % K;
  wt[k * N + n] = (bf16)w[i];
}
// conv weights with spatial-major k: wt[(sp*CI+c)*CO + n] = w[n][c][sp]
__global__ void prep_convw(const float* __restrict__ w, bf16* __restrict__ wt,
                           int CO, int CI, int KHW) {
  int i = blockIdx.x * 256 + threadIdx.x;
  if (i >= CO * CI * KHW) return;
  int k = i / CO, n = i % CO;
  int sp = k / CI, c = k % CI;
  wt[i] = (bf16)w[(n * CI + c) * KHW + sp];
}
// fc weights permuted for NHWC-flat input: k2 = (y*7+x)*64 + c
__global__ void prep_fc(const float* __restrict__ w, bf16* __restrict__ wt) {
  int i = blockIdx.x * 256 + threadIdx.x;
  if (i >= 3136 * 512) return;
  int k2 = i / 512, n = i % 512;
  int sp = k2 >> 6, c = k2 & 63;
  wt[i] = (bf16)w[n * 3136 + c * 49 + sp];
}

__global__ void init_state(const float* __restrict__ h0in, const float* __restrict__ c0in,
                           float* __restrict__ h0, float* __restrict__ cbuf) {
  int i = blockIdx.x * 256 + threadIdx.x;
  if (i < 8192) { h0[i] = h0in[i]; cbuf[i] = c0in[i]; }
}

// ---------------- implicit-GEMM conv (bf16 WMMA) ----------------
// NHWC=false: f32 NCHW input (conv1), c-major k, elementwise gather.
// NHWC=true : bf16 NHWC input, spatial-major k, async 64B-run A fill.
template<int CI,int HI,int WI,int KH,int KW,int S,int CO,int HO,int WO,bool NHWC>
__global__ __launch_bounds__(64*(CO/16)) void conv_wmma(
    const void* __restrict__ in_, const bf16* __restrict__ wB,
    const float* __restrict__ bias, bf16* __restrict__ out)
{
  constexpr int K  = CI * KH * KW;
  constexpr int NT = 64 * (CO / 16);
  constexpr int LDA = 40, LDK = 40;
  __shared__ __align__(16) bf16 As[32 * LDA];
  __shared__ __align__(16) bf16 Bs[CO * LDK];
  const int tid = threadIdx.x, lane = tid & 31, w = tid >> 5;
  const int wm = w & 1, wn = w >> 1;
  const int m0 = blockIdx.x * 32;
  const float* inF = (const float*)in_;
  const bf16*  inB = (const bf16*)in_;

  // per-thread row decode, hoisted out of the K loop (async path: NT=256)
  const int arow = tid >> 3, aq = tid & 7;
  int a_img = 0, a_oy = 0, a_ox = 0;
  if (NHWC) {
    int m = m0 + arow;
    a_img = m / (HO * WO);
    int rem = m % (HO * WO);
    a_oy = rem / WO; a_ox = rem % WO;
  }

  v8f acc = {};
  for (int kc = 0; kc < K; kc += 32) {
    if (NHWC) {
      // one contiguous 64B run per row: k = sp*CI + c, chunk stays inside one sp (CI>=32)
      int sp = kc / CI, c0 = kc % CI;
      int kh = sp / KW, kw = sp % KW;
      const bf16* g = inB + ((size_t)((a_img * HI + a_oy * S + kh) * WI) + (a_ox * S + kw)) * CI
                          + c0 + aq * 4;
      async_copy_b64(&As[arow * LDA + aq * 4], g);
    } else {
      for (int idx = tid; idx < 32 * 32; idx += NT) {
        int row = idx >> 5, kk = idx & 31;
        int m = m0 + row;
        int img = m / (HO * WO), rem = m % (HO * WO);
        int oy = rem / WO, ox = rem % WO;
        int k = kc + kk;
        int c = k / (KH * KW), r2 = k % (KH * KW);
        int kh = r2 / KW, kw = r2 % KW;
        As[row * LDA + kk] =
            (bf16)(inF[((img * CI + c) * HI + oy * S + kh) * WI + ox * S + kw] * 0.003921568859f);
      }
    }
    for (int idx = tid; idx < 32 * CO; idx += NT) {   // B tile, transposed [N][K]
      int k = idx / CO, n = idx % CO;
      Bs[n * LDK + k] = wB[(kc + k) * CO + n];
    }
    if (NHWC) async_wait0();
    __syncthreads();
    v16bf a = frag_a(&As[(wm * 16) * LDA], LDA, lane);
    v16bf b = frag_bt(&Bs[(wn * 16) * LDK], LDK, lane);
    acc = __builtin_amdgcn_wmma_f32_16x16x32_bf16(false, a, false, b, (short)0, acc, false, false);
    __syncthreads();
  }
  int nl = lane & 15, mb = (lane >> 4) * 8;
  int n = wn * 16 + nl;
  float bv = bias[n];
#pragma unroll
  for (int i = 0; i < 8; ++i) {
    int m = m0 + wm * 16 + mb + i;
    float vv = acc[i] + bv;
    vv = vv > 0.f ? vv : 0.f;
    out[(size_t)m * CO + n] = (bf16)vv;   // NHWC row-major [M][CO]
  }
}

// ---------------- generic bf16 GEMM: C = A[MxK] * B[KxN] + epilogue ----------------
// MODE 0: out bf16 = relu(acc + bias0[n])        (FC)
// MODE 1: out f32  = acc + bias0[n] + bias1[n]   (LSTM x-gates)
template<int MODE>
__global__ __launch_bounds__(256) void gemm_wmma(
    const bf16* __restrict__ A, const bf16* __restrict__ B,
    const float* __restrict__ bias0, const float* __restrict__ bias1,
    void* __restrict__ out, int M, int N, int K)
{
  constexpr int LDA = 40, LDK = 40;
  __shared__ __align__(16) bf16 As[32 * LDA];
  __shared__ __align__(16) bf16 Bs[64 * LDK];
  const int tid = threadIdx.x, lane = tid & 31, w = tid >> 5;
  const int wm = w & 1, wn = w >> 1;               // 2x4 waves -> 32x64 tile
  const int m0 = blockIdx.x * 32, n0 = blockIdx.y * 64;
  const int arow = tid >> 3, aq = tid & 7;
  const bf16* arowp = A + (size_t)(m0 + arow) * K + aq * 4;
  v8f acc = {};
  for (int kc = 0; kc < K; kc += 32) {
    async_copy_b64(&As[arow * LDA + aq * 4], arowp + kc);      // A panel: async 8B/lane
    for (int idx = tid; idx < 2048; idx += 256) {              // B panel, transposed [N][K]
      int k = idx >> 6, n = idx & 63;
      Bs[n * LDK + k] = B[(size_t)(kc + k) * N + n0 + n];
    }
    if (kc + 32 < K)  // gfx1250 global_prefetch path for next B panel
      __builtin_prefetch(&B[(size_t)(kc + 32) * N + n0 + (tid & 63)], 0, 3);
    async_wait0();
    __syncthreads();
    v16bf a = frag_a(&As[(wm * 16) * LDA], LDA, lane);
    v16bf b = frag_bt(&Bs[(wn * 16) * LDK], LDK, lane);
    acc = __builtin_amdgcn_wmma_f32_16x16x32_bf16(false, a, false, b, (short)0, acc, false, false);
    __syncthreads();
  }
  int nl = lane & 15, mb = (lane >> 4) * 8;
  int n = n0 + wn * 16 + nl;
#pragma unroll
  for (int i = 0; i < 8; ++i) {
    int m = m0 + wm * 16 + mb + i;
    float vv = acc[i];
    if (MODE == 0) {
      vv += bias0[n];
      vv = vv > 0.f ? vv : 0.f;
      ((bf16*)out)[(size_t)m * N + n] = (bf16)vv;
    } else {
      vv += bias0[n] + bias1[n];
      ((float*)out)[(size_t)m * N + n] = vv;
    }
  }
}

// ---------------- one LSTM timestep ----------------
__global__ __launch_bounds__(256) void lstm_step(
    const float* __restrict__ hprev, const float* __restrict__ done,
    const bf16* __restrict__ whhT, const float* __restrict__ gatesx,
    float* __restrict__ cbuf, float* __restrict__ hs, int t)
{
  constexpr int LDH = 264, LDK = 264;
  __shared__ __align__(16) bf16  hm[32 * LDH];
  __shared__ __align__(16) bf16  Bs[64 * LDK];     // transposed [col][K]
  __shared__ float gsum[4][32][16];
  const int tid = threadIdx.x, lane = tid & 31, w = tid >> 5;
  const int wm = w & 1, g = w >> 1;                // wave -> (M-tile, gate)
  const int jbase = blockIdx.x * 16;
  for (int idx = tid; idx < 32 * 256; idx += 256) {   // masked h -> bf16 LDS
    int b = idx >> 8, k = idx & 255;
    float mval = 1.0f - done[t * 32 + b];
    hm[b * LDH + k] = (bf16)(mval * hprev[b * 256 + k]);
  }
  for (int idx = tid; idx < 64 * 256; idx += 256) {   // W_hh cols for 4 gates
    int k = idx >> 6, c = idx & 63;                   // coalesced over c
    int gg = c >> 4, jc = c & 15;
    Bs[c * LDK + k] = whhT[k * 1024 + gg * 256 + jbase + jc];
  }
  __syncthreads();
  v8f acc = {};
#pragma unroll
  for (int kc = 0; kc < 256; kc += 32) {
    v16bf a = frag_a(&hm[(wm * 16) * LDH + kc], LDH, lane);
    v16bf b = frag_bt(&Bs[(g * 16) * LDK + kc], LDK, lane);
    acc = __builtin_amdgcn_wmma_f32_16x16x32_bf16(false, a, false, b, (short)0, acc, false, false);
  }
  int nl = lane & 15, mb = (lane >> 4) * 8;
#pragma unroll
  for (int i = 0; i < 8; ++i) {
    int b = wm * 16 + mb + i;
    gsum[g][b][nl] = acc[i] + gatesx[(size_t)(t * 32 + b) * 1024 + g * 256 + jbase + nl];
  }
  __syncthreads();
  for (int idx = tid; idx < 512; idx += 256) {
    int b = idx >> 4, jc = idx & 15;
    int j = jbase + jc;
    float ig = 1.f / (1.f + __expf(-gsum[0][b][jc]));
    float fg = 1.f / (1.f + __expf(-gsum[1][b][jc]));
    float gg = tanhf(gsum[2][b][jc]);
    float og = 1.f / (1.f + __expf(-gsum[3][b][jc]));
    float cp = (1.0f - done[t * 32 + b]) * cbuf[b * 256 + j];
    float cn = fg * cp + ig * gg;
    cbuf[b * 256 + j] = cn;
    hs[(size_t)(t * 32 + b) * 256 + j] = og * tanhf(cn);
  }
}

// ---------------- heads + finalize ----------------
__global__ __launch_bounds__(32) void heads_kernel(
    const float* __restrict__ hs, const float* __restrict__ polw,
    const float* __restrict__ polb, const float* __restrict__ valw,
    const float* __restrict__ valb, float* __restrict__ out)
{
  int m = blockIdx.x, lane = threadIdx.x;
  float a0=0,a1=0,a2=0,a3=0,a4=0,a5=0,av=0;
#pragma unroll
  for (int kk = 0; kk < 8; ++kk) {
    int k = lane + kk * 32;
    float hv = hs[(size_t)m * 256 + k];
    a0 += hv * polw[0*256 + k];  a1 += hv * polw[1*256 + k];
    a2 += hv * polw[2*256 + k];  a3 += hv * polw[3*256 + k];
    a4 += hv * polw[4*256 + k];  a5 += hv * polw[5*256 + k];
    av += hv * valw[k];
  }
#pragma unroll
  for (int off = 16; off > 0; off >>= 1) {
    a0 += __shfl_xor(a0, off); a1 += __shfl_xor(a1, off);
    a2 += __shfl_xor(a2, off); a3 += __shfl_xor(a3, off);
    a4 += __shfl_xor(a4, off); a5 += __shfl_xor(a5, off);
    av += __shfl_xor(av, off);
  }
  if (lane == 0) {
    out[m*6+0] = a0 + polb[0]; out[m*6+1] = a1 + polb[1];
    out[m*6+2] = a2 + polb[2]; out[m*6+3] = a3 + polb[3];
    out[m*6+4] = a4 + polb[4]; out[m*6+5] = a5 + polb[5];
    out[4096*6 + m] = av + valb[0];
  }
}

__global__ void finalize_kernel(const float* __restrict__ hs, const float* __restrict__ cbuf,
                                float* __restrict__ out) {
  int i = blockIdx.x * 256 + threadIdx.x;
  if (i < 8192) {
    out[28672 + i]        = hs[127 * 32 * 256 + i];  // state_hidden = h_T
    out[28672 + 8192 + i] = cbuf[i];                 // state_cell   = c_T
  }
}

extern "C" void kernel_launch(void* const* d_in, const int* in_sizes, int n_in,
                              void* d_out, int out_size, void* d_ws, size_t ws_size,
                              hipStream_t stream) {
  (void)in_sizes; (void)n_in; (void)out_size; (void)ws_size;
  const float* image   = (const float*)d_in[0];
  const float* done    = (const float*)d_in[1];
  const float* hidden0 = (const float*)d_in[2];
  const float* cell0   = (const float*)d_in[3];
  const float* c1_w = (const float*)d_in[4];
  const float* c1_b = (const float*)d_in[5];
  const float* c2_w = (const float*)d_in[6];
  const float* c2_b = (const float*)d_in[7];
  const float* c3_w = (const float*)d_in[8];
  const float* c3_b = (const float*)d_in[9];
  const float* fc_w = (const float*)d_in[10];
  const float* fc_b = (const float*)d_in[11];
  const float* w_ih = (const float*)d_in[12];
  const float* w_hh = (const float*)d_in[13];
  const float* b_ih = (const float*)d_in[14];
  const float* b_hh = (const float*)d_in[15];
  const float* pol_w = (const float*)d_in[16];
  const float* pol_b = (const float*)d_in[17];
  const float* val_w = (const float*)d_in[18];
  const float* val_b = (const float*)d_in[19];
  float* out = (float*)d_out;

  char* ws = (char*)d_ws;
  size_t off = 0;
  auto alloc = [&](size_t bytes) {
    void* p = ws + off;
    off = (off + bytes + 255) & ~(size_t)255;
    return p;
  };
  bf16*  regionA = (bf16*)alloc((size_t)4096 * 32 * 400 * 2);  // conv1 out; aliased by conv3/flat
  bf16*  conv2o  = (bf16*)alloc((size_t)4096 * 64 * 81 * 2);
  bf16*  wB1  = (bf16*)alloc((size_t)192 * 32 * 2);
  bf16*  wB2  = (bf16*)alloc((size_t)512 * 64 * 2);
  bf16*  wB3  = (bf16*)alloc((size_t)576 * 64 * 2);
  bf16*  fcT  = (bf16*)alloc((size_t)3136 * 512 * 2);
  bf16*  ihT  = (bf16*)alloc((size_t)512 * 1024 * 2);
  bf16*  hhT  = (bf16*)alloc((size_t)256 * 1024 * 2);
  bf16*  feats  = (bf16*)alloc((size_t)4096 * 512 * 2);
  float* gatesx = (float*)alloc((size_t)4096 * 1024 * 4);
  float* hsbuf  = (float*)alloc((size_t)4096 * 256 * 4);
  float* h0   = (float*)alloc(8192 * 4);
  float* cbuf = (float*)alloc(8192 * 4);

  auto g256 = [](int n) { return (n + 255) / 256; };
  prep_wt<<<g256(192*32),     256, 0, stream>>>(c1_w, wB1, 32, 192);     // conv1: c-major k
  prep_convw<<<g256(512*64),  256, 0, stream>>>(c2_w, wB2, 64, 32, 16);  // spatial-major k
  prep_convw<<<g256(576*64),  256, 0, stream>>>(c3_w, wB3, 64, 64, 9);
  prep_fc<<<g256(3136*512),   256, 0, stream>>>(fc_w, fcT);              // NHWC-permuted K
  prep_wt<<<g256(512*1024),   256, 0, stream>>>(w_ih, ihT, 1024, 512);
  prep_wt<<<g256(256*1024),   256, 0, stream>>>(w_hh, hhT, 1024, 256);
  init_state<<<32, 256, 0, stream>>>(hidden0, cell0, h0, cbuf);

  // conv1: M=4096*400, N=32, K=192 (f32 NCHW in -> bf16 NHWC out)
  conv_wmma<3,84,84,8,8,4,32,20,20,false><<<51200, 128, 0, stream>>>(image, wB1, c1_b, regionA);
  // conv2: M=4096*81, N=64, K=512 (NHWC, async A fill)
  conv_wmma<32,20,20,4,4,2,64,9,9,true><<<10368, 256, 0, stream>>>(regionA, wB2, c2_b, conv2o);
  // conv3: M=4096*49, N=64, K=576 -> NHWC flat [4096][3136]
  conv_wmma<64,9,9,3,3,1,64,7,7,true><<<6272, 256, 0, stream>>>(conv2o, wB3, c3_b, regionA);
  // FC: 4096x512x3136 -> relu bf16 feats
  gemm_wmma<0><<<dim3(128, 8), 256, 0, stream>>>(regionA, fcT, fc_b, nullptr, feats, 4096, 512, 3136);
  // LSTM x-gates hoisted: 4096x1024x512 (+b_ih+b_hh) -> f32
  gemm_wmma<1><<<dim3(128, 16), 256, 0, stream>>>(feats, ihT, b_ih, b_hh, gatesx, 4096, 1024, 512);
  // recurrence: 128 small WMMA steps
  for (int t = 0; t < 128; ++t) {
    const float* hp = (t == 0) ? h0 : (hsbuf + (size_t)(t - 1) * 32 * 256);
    lstm_step<<<16, 256, 0, stream>>>(hp, done, hhT, gatesx, cbuf, hsbuf, t);
  }
  heads_kernel<<<4096, 32, 0, stream>>>(hsbuf, pol_w, pol_b, val_w, val_b, out);
  finalize_kernel<<<32, 256, 0, stream>>>(hsbuf, cbuf, out);
}
